// GNNForBERT_81827716924083
// MI455X (gfx1250) — matrix operand order repaired and checked by reading, not compiled, not run on hardware
//
#include <hip/hip_runtime.h>

typedef unsigned int   u32;
typedef unsigned short u16;
typedef unsigned long long u64;

typedef __attribute__((ext_vector_type(16))) __bf16 v16bf;
typedef __attribute__((ext_vector_type(8)))  float  v8f;

union Frag { u32 u[8]; v16bf v; };

__device__ __forceinline__ u16 f2bf(float f) {
    u32 u = __float_as_uint(f);
    u32 r = (u + 0x7FFFu + ((u >> 16) & 1u)) >> 16;
    return (u16)r;
}

// CDNA5 async global->LDS DMA (ASYNCcnt-tracked), GVS addressing:
//   LDS address from per-lane VDST; global address = SGPR base + per-lane VGPR offset
__device__ __forceinline__ void async_ld16(u32 lds_addr, u64 gbase, u32 goff) {
    asm volatile("global_load_async_to_lds_b128 %0, %1, %2"
                 :: "v"(lds_addr), "v"(goff), "s"(gbase) : "memory");
}

// Fused sync: wait own async fills + own LDS ops, then full workgroup barrier.
// Returns an always-zero SGPR token folded into subsequent LDS addresses ->
// true data dependence, so fragment ds_loads cannot be hoisted above the
// wait/barrier (hazard observed in round 2 asm, fixed since round 3).
__device__ __forceinline__ u32 sync_wait_async() {
    u32 z;
    asm volatile(
        "s_wait_asynccnt 0x0\n\t"
        "s_wait_dscnt 0x0\n\t"
        "s_barrier_signal -1\n\t"
        "s_barrier_wait -1\n\t"
        "s_mov_b32 %0, 0"
        : "=s"(z) : : "memory");
    __builtin_assume((z & 63u) == 0u);   // keep LDS alignment provable
    return z;
}

#define N_NODES 8192
#define DIM     768
#define HID     768
#define HEADS   4
#define KNN     5
#define EDGES   (N_NODES * 6)

// ---------------------------------------------------------------------------
// K0: per-row L2 norm; emit bf16 normalized rows (similarity) + raw bf16 rows
// ---------------------------------------------------------------------------
__global__ __launch_bounds__(256) void norm_convert_kernel(
    const float* __restrict__ x, u16* __restrict__ xn, u16* __restrict__ xraw)
{
    int wid  = blockIdx.x * 8 + (threadIdx.x >> 5);
    int lane = threadIdx.x & 31;
    if (wid >= N_NODES) return;
    const float* row = x + (size_t)wid * DIM;
    float ss = 0.f;
    for (int c = lane; c < DIM; c += 32) { float v = row[c]; ss += v * v; }
    for (int o = 16; o; o >>= 1) ss += __shfl_xor(ss, o, 32);
    float rinv = rsqrtf(ss);
    for (int c = lane; c < DIM; c += 32) {
        float v = row[c];
        xraw[(size_t)wid * DIM + c] = f2bf(v);
        xn  [(size_t)wid * DIM + c] = f2bf(v * rinv);
    }
}

// ---------------------------------------------------------------------------
// Kp: pack f32 weight [K,N] -> bf16 K-pair interleaved u32 [K/2, N]
// ---------------------------------------------------------------------------
__global__ __launch_bounds__(256) void pack_weight_kernel(
    const float* __restrict__ W, u32* __restrict__ Wi, int K, int N)
{
    int idx = blockIdx.x * 256 + threadIdx.x;
    int total = (K / 2) * N;
    if (idx >= total) return;
    int kp = idx / N, n = idx % N;
    float a = W[(size_t)(2 * kp) * N + n];
    float b = W[(size_t)(2 * kp + 1) * N + n];
    Wi[idx] = (u32)f2bf(a) | ((u32)f2bf(b) << 16);
}

// ---------------------------------------------------------------------------
// K1: fused cosine-similarity GEMM + per-row top-6 (drop max == self) -> nbr
// ---------------------------------------------------------------------------
__global__ __launch_bounds__(256) void sim_topk_kernel(
    const u16* __restrict__ xn, int* __restrict__ nbr)
{
    __shared__ __align__(16) u16 As[16][776];       // 16 x 768 (+8 pad), 24832B
    __shared__ __align__(16) u16 Bs[2][128][72];    // 2 x 128 x 64 (+8 pad), 36864B

    const int t    = threadIdx.x;
    const int w    = t >> 5;
    const int lane = t & 31;
    const int row0 = blockIdx.x * 16;
    const int r16  = lane & 15;
    const int khi  = lane >> 4;
    const u64 gx   = (u64)xn;

    // async-stage A slab (16 x 768 halfs) once
#pragma unroll
    for (int i = 0; i < 6; ++i) {
        int u = t + i * 256;
        int r = u / 96, c8 = (u % 96) * 8;
        async_ld16((u32)(uintptr_t)&As[r][c8], gx, (u32)(((row0 + r) * DIM + c8) * 2));
    }

    float tv[6]; int tc[6];
#pragma unroll
    for (int q = 0; q < 6; ++q) { tv[q] = -3.0e38f; tc[q] = -1; }

    for (int jb = 0; jb < 64; ++jb) {
        v8f acc = {0.f, 0.f, 0.f, 0.f, 0.f, 0.f, 0.f, 0.f};
        // prologue stage of chunk 0 into buf 0
#pragma unroll
        for (int i = 0; i < 4; ++i) {
            int u = t + i * 256;
            int r = u >> 3, c8 = (u & 7) * 8;
            async_ld16((u32)(uintptr_t)&Bs[0][r][c8], gx,
                       (u32)(((jb * 128 + r) * DIM + c8) * 2));
        }
        for (int kc = 0; kc < 12; ++kc) {
            u32 tok = sync_wait_async();
            const u16* AsT = (const u16*)((const char*)&As[0][0] + tok);
            const u16* BsT = (const u16*)((const char*)&Bs[0][0][0] + tok);
            if (kc < 11) {   // prefetch next chunk into other buffer
                int nb = (kc + 1) & 1;
#pragma unroll
                for (int i = 0; i < 4; ++i) {
                    int u = t + i * 256;
                    int r = u >> 3, c8 = (u & 7) * 8;
                    async_ld16((u32)(uintptr_t)&Bs[nb][r][c8], gx,
                               (u32)(((jb * 128 + r) * DIM + (kc + 1) * 64 + c8) * 2));
                }
            }
            const int buf = kc & 1;
#pragma unroll
            for (int ks = 0; ks < 2; ++ks) {
                Frag a, b;
                int kbA = kc * 64 + ks * 32;
#pragma unroll
                for (int v = 0; v < 8; ++v) {
                    int koA = (v < 4) ? (2 * v + 8 * khi) : (16 + 2 * (v - 4) + 8 * khi);
                    a.u[v] = *(const u32*)&AsT[r16 * 776 + kbA + koA];
                    b.u[v] = *(const u32*)&BsT[buf * (128 * 72) + (w * 16 + r16) * 72 +
                                               ks * 32 + koA];
                }
                acc = __builtin_amdgcn_wmma_f32_16x16x32_bf16(
                    false, a.v, false, b.v, (short)0, acc, false, false);
            }
        }
        // tile -> LDS (overlay buf0; idle, last compute used buf1); per-wave region
        float* Ct = (float*)&Bs[0][0][0];   // [8][16][17]
#pragma unroll
        for (int vg = 0; vg < 8; ++vg) {
            int m = vg + 8 * khi;
            Ct[(w * 16 + m) * 17 + r16] = acc[vg];
        }
        if (lane < 16) {
            int r = lane;
#pragma unroll
            for (int n = 0; n < 16; ++n) {
                float v = Ct[(w * 16 + r) * 17 + n];
                int   c = jb * 128 + w * 16 + n;
#pragma unroll
                for (int q = 0; q < 6; ++q) {
                    bool sw = (v > tv[q]) || (v == tv[q] && c < tc[q]);
                    if (sw) { float fv = tv[q]; int fc = tc[q]; tv[q] = v; tc[q] = c; v = fv; c = fc; }
                }
            }
        }
        (void)sync_wait_async();   // protect Ct reads from next jb's async staging
    }
    // merge 8 waves' candidates (overlay As)
    float* mv = (float*)&As[0][0];
    int*   mc = (int*)((char*)&As[0][0] + 8 * 16 * 6 * 4);
    if (lane < 16) {
#pragma unroll
        for (int q = 0; q < 6; ++q) {
            mv[(w * 16 + lane) * 6 + q] = tv[q];
            mc[(w * 16 + lane) * 6 + q] = tc[q];
        }
    }
    u32 tok2 = sync_wait_async();
    const float* mvR = (const float*)((const char*)&As[0][0] + tok2);
    const int*   mcR = (const int*)((const char*)&As[0][0] + 8 * 16 * 6 * 4 + tok2);
    if (t < 16) {
        int r = t;
        float cv[48]; int cc[48];
        for (int ww = 0; ww < 8; ++ww)
            for (int q = 0; q < 6; ++q) {
                cv[ww * 6 + q] = mvR[(ww * 16 + r) * 6 + q];
                cc[ww * 6 + q] = mcR[(ww * 16 + r) * 6 + q];
            }
        for (int s = 0; s < 6; ++s) {
            int bi = 0; float bv = cv[0]; int bc = cc[0];
            for (int q = 1; q < 48; ++q) {
                bool bt = (cv[q] > bv) || (cv[q] == bv && cc[q] < bc);
                if (bt) { bv = cv[q]; bc = cc[q]; bi = q; }
            }
            cv[bi] = -3.4e38f;
            if (s >= 1) nbr[(row0 + r) * KNN + (s - 1)] = bc;  // drop top-1 (self)
        }
    }
}

// ---------------------------------------------------------------------------
// K2: bf16 WMMA GEMM  C[M,N] = A[M,K](bf16) @ B(packed interleaved bf16)
// ---------------------------------------------------------------------------
__global__ __launch_bounds__(256) void gemm_bf16_kernel(
    const u16* __restrict__ A, const u32* __restrict__ Bi,
    const float* __restrict__ bias, float* __restrict__ outF, u16* __restrict__ outBf,
    int M, int N, int K, int relu)
{
    __shared__ __align__(16) u16 As[2][128][40];   // 2 x 128 x 32 (+8 pad), 20480B
    __shared__ __align__(16) u32 Bsi[2][16][64];   // K-pair interleaved, 8192B

    const int t    = threadIdx.x;
    const int w    = t >> 5;
    const int lane = t & 31;
    const int wm   = w & 3, wn = w >> 2;
    const int m0   = blockIdx.y * 128;
    const int n0   = blockIdx.x * 64;
    const int r16  = lane & 15, khi = lane >> 4;
    const u64 ga   = (u64)A;
    const u64 gb   = (u64)Bi;

    v8f zero = {0.f, 0.f, 0.f, 0.f, 0.f, 0.f, 0.f, 0.f};
    v8f acc[2][2];
    acc[0][0] = zero; acc[0][1] = zero; acc[1][0] = zero; acc[1][1] = zero;

    auto stage = [&](int buf, int kc) {
#pragma unroll
        for (int i = 0; i < 2; ++i) {                  // A: 128 x 32 halfs
            int u = t + i * 256;
            int r = u >> 2, c8 = (u & 3) * 8;
            async_ld16((u32)(uintptr_t)&As[buf][r][c8], ga,
                       (u32)(((m0 + r) * K + kc + c8) * 2));
        }
        {                                              // B: 16 x 64 u32 (one unit/thread)
            int r = t >> 4, c4 = (t & 15) * 4;
            async_ld16((u32)(uintptr_t)&Bsi[buf][r][c4], gb,
                       (u32)(((kc / 2 + r) * N + n0 + c4) * 4));
        }
    };

    stage(0, 0);
    const int nk = K / 32;
    for (int ki = 0; ki < nk; ++ki) {
        u32 tok = sync_wait_async();
        const u16* AsT = (const u16*)((const char*)&As[0][0][0] + tok);
        const u32* BsT = (const u32*)((const char*)&Bsi[0][0][0] + tok);
        if (ki + 1 < nk) stage((ki + 1) & 1, (ki + 1) * 32);
        const int buf = ki & 1;
        Frag a[2], b[2];
#pragma unroll
        for (int tm = 0; tm < 2; ++tm) {
            int ar = wm * 32 + tm * 16 + r16;
#pragma unroll
            for (int v = 0; v < 8; ++v) {
                int koA = 16 * khi + 2 * v;
                a[tm].u[v] = *(const u32*)&AsT[buf * (128 * 40) + ar * 40 + koA];
            }
        }
#pragma unroll
        for (int tn = 0; tn < 2; ++tn) {
#pragma unroll
            for (int v = 0; v < 8; ++v)
                b[tn].u[v] = BsT[buf * (16 * 64) + (8 * khi + v) * 64 +
                                 wn * 32 + tn * 16 + r16];
        }
#pragma unroll
        for (int tm = 0; tm < 2; ++tm)
#pragma unroll
            for (int tn = 0; tn < 2; ++tn)
                acc[tm][tn] = __builtin_amdgcn_wmma_f32_16x16x32_bf16(
                    false, a[tm].v, false, b[tn].v, (short)0, acc[tm][tn], false, false);
    }

#pragma unroll
    for (int tm = 0; tm < 2; ++tm)
#pragma unroll
        for (int tn = 0; tn < 2; ++tn) {
            int n = n0 + wn * 32 + tn * 16 + r16;
            if (n < N) {
                float bs = bias ? bias[n] : 0.f;
#pragma unroll
                for (int vg = 0; vg < 8; ++vg) {
                    int m = m0 + wm * 32 + tm * 16 + vg + 8 * khi;
                    float v = acc[tm][tn][vg] + bs;
                    if (relu) v = fmaxf(v, 0.f);
                    if (outF)  outF[(size_t)m * N + n] = v;
                    if (outBf) outBf[(size_t)m * N + n] = f2bf(v);
                }
            }
        }
}

// ---------------------------------------------------------------------------
// K3: attention logits  al_s[i,h] = <h[i,h,:], a_src[h,:]>, same for a_dst
// ---------------------------------------------------------------------------
__global__ __launch_bounds__(256) void attn_logits_kernel(
    const float* __restrict__ h, const float* __restrict__ asrc,
    const float* __restrict__ adst, float* __restrict__ als, float* __restrict__ ald,
    int H, int C)
{
    int wid  = blockIdx.x * 8 + (threadIdx.x >> 5);
    int lane = threadIdx.x & 31;
    if (wid >= N_NODES * H) return;
    int i = wid / H, hh = wid % H;
    const float* hp = h + (size_t)i * H * C + hh * C;
    const float* sp = asrc + (size_t)hh * C;
    const float* dp = adst + (size_t)hh * C;
    float s = 0.f, d = 0.f;
    for (int c = lane; c < C; c += 32) { float v = hp[c]; s += v * sp[c]; d += v * dp[c]; }
    for (int o = 16; o; o >>= 1) { s += __shfl_xor(s, o, 32); d += __shfl_xor(d, o, 32); }
    if (lane == 0) { als[wid] = s; ald[wid] = d; }
}

// ---------------------------------------------------------------------------
// K4: per-edge weight = exp(leaky_relu(als[src]+ald[dst])); atomic den[dst]
// ---------------------------------------------------------------------------
__global__ __launch_bounds__(256) void edge_weights_kernel(
    const int* __restrict__ nbr, const float* __restrict__ als,
    const float* __restrict__ ald, float* __restrict__ ew, float* __restrict__ den, int H)
{
    int eid = blockIdx.x * blockDim.x + threadIdx.x;
    if (eid >= EDGES) return;
    int i = eid / 6, tt = eid % 6;
    int j = (tt < 5) ? nbr[i * KNN + tt] : i;
    for (int hh = 0; hh < H; ++hh) {
        float e = als[i * H + hh] + ald[j * H + hh];
        e = (e > 0.f) ? e : 0.2f * e;
        float wv = __expf(e);
        ew[(size_t)eid * H + hh] = wv;
        __hip_atomic_fetch_add(&den[j * H + hh], wv, __ATOMIC_RELAXED, __HIP_MEMORY_SCOPE_AGENT);
    }
}

// ---------------------------------------------------------------------------
// K5: scatter  acc[dst] += coef * h[src]   (block per edge)
// ---------------------------------------------------------------------------
__global__ __launch_bounds__(256) void scatter_kernel(
    const int* __restrict__ nbr, const float* __restrict__ ew,
    const float* __restrict__ den, const float* __restrict__ h,
    float* __restrict__ acc, int H, int C)
{
    int eid = blockIdx.x;
    int i = eid / 6, tt = eid % 6;
    int j = (tt < 5) ? nbr[i * KNN + tt] : i;
    float coef[4];
    for (int hh = 0; hh < H; ++hh)
        coef[hh] = ew[(size_t)eid * H + hh] / den[j * H + hh];
    int HC = H * C;
    const float* hp = h + (size_t)i * HC;
    float* ap = acc + (size_t)j * HC;
    for (int f = threadIdx.x; f < HC; f += 256) {
        int hh = f / C;
        __hip_atomic_fetch_add(&ap[f], coef[hh] * hp[f],
                               __ATOMIC_RELAXED, __HIP_MEMORY_SCOPE_AGENT);
    }
}

// ---------------------------------------------------------------------------
// K6: h = relu(acc + bias[col]) + res  -> bf16
// ---------------------------------------------------------------------------
__global__ __launch_bounds__(256) void epilogue_kernel(
    const float* __restrict__ acc, const float* __restrict__ bias,
    const float* __restrict__ res, u16* __restrict__ outBf, int total, int cols)
{
    for (int idx = blockIdx.x * blockDim.x + threadIdx.x; idx < total;
         idx += gridDim.x * blockDim.x) {
        int c = idx % cols;
        float v = fmaxf(acc[idx] + bias[c], 0.f) + res[idx];
        outBf[idx] = f2bf(v);
    }
}

// ---------------------------------------------------------------------------
// K8: in-place row L2 normalize
// ---------------------------------------------------------------------------
__global__ __launch_bounds__(256) void l2norm_kernel(float* __restrict__ y, int C)
{
    int wid  = blockIdx.x * 8 + (threadIdx.x >> 5);
    int lane = threadIdx.x & 31;
    if (wid >= N_NODES) return;
    float* row = y + (size_t)wid * C;
    float ss = 0.f;
    for (int c = lane; c < C; c += 32) { float v = row[c]; ss += v * v; }
    for (int o = 16; o; o >>= 1) ss += __shfl_xor(ss, o, 32);
    float rinv = rsqrtf(ss);
    for (int c = lane; c < C; c += 32) row[c] *= rinv;
}

// ---------------------------------------------------------------------------
extern "C" void kernel_launch(void* const* d_in, const int* in_sizes, int n_in,
                              void* d_out, int out_size, void* d_ws, size_t ws_size,
                              hipStream_t stream)
{
    const float* x    = (const float*)d_in[0];
    const float* W1   = (const float*)d_in[1];
    const float* as1  = (const float*)d_in[2];
    const float* ad1  = (const float*)d_in[3];
    const float* b1   = (const float*)d_in[4];
    const float* W2   = (const float*)d_in[5];
    const float* as2  = (const float*)d_in[6];
    const float* ad2  = (const float*)d_in[7];
    const float* b2   = (const float*)d_in[8];
    const float* Wr1  = (const float*)d_in[9];
    const float* br1  = (const float*)d_in[10];
    const float* Wr2  = (const float*)d_in[11];
    const float* br2  = (const float*)d_in[12];
    const float* Wh1  = (const float*)d_in[13];
    const float* bh1  = (const float*)d_in[14];
    const float* Wh2  = (const float*)d_in[15];
    const float* bh2  = (const float*)d_in[16];
    const float* Wf   = (const float*)d_in[17];
    const float* bf   = (const float*)d_in[18];

    char* ws = (char*)d_ws;
    size_t off = 0;
    auto alloc = [&](size_t bytes) -> char* {
        char* p = ws + off;
        off += (bytes + 255) & ~(size_t)255;
        return p;
    };

    int*   nbr   = (int*)  alloc((size_t)N_NODES * KNN * 4);
    u16*   xnbf  = (u16*)  alloc((size_t)N_NODES * DIM * 2);     // reused as tbf
    u16*   xbf   = (u16*)  alloc((size_t)N_NODES * DIM * 2);     // reused as featbf
    u32*   W1p   = (u32*)  alloc((size_t)(DIM/2) * 3072 * 4);
    u32*   Wr1p  = (u32*)  alloc((size_t)(DIM/2) * 3072 * 4);
    u32*   W2p   = (u32*)  alloc((size_t)(3072/2) * DIM * 4);
    u32*   Wr2p  = (u32*)  alloc((size_t)(3072/2) * DIM * 4);
    u32*   Wh1p  = (u32*)  alloc((size_t)(DIM/2) * HID * 4);
    u32*   Wh2p  = (u32*)  alloc((size_t)(DIM/2) * 128 * 4);
    u32*   Wfp   = (u32*)  alloc((size_t)(DIM/2) * 5 * 4);
    float* bufA  = (float*)alloc((size_t)N_NODES * 3072 * 4);    // hlin1/xres1/hlin2/xres2
    float* bufB  = (float*)alloc((size_t)N_NODES * 3072 * 4);    // gat accumulators
    u16*   hbf   = (u16*)  alloc((size_t)N_NODES * 3072 * 2);
    float* als1  = (float*)alloc((size_t)N_NODES * HEADS * 4);
    float* ald1  = (float*)alloc((size_t)N_NODES * HEADS * 4);
    float* den1  = (float*)alloc((size_t)N_NODES * HEADS * 4);
    float* ew1   = (float*)alloc((size_t)EDGES * HEADS * 4);
    float* als2  = (float*)alloc((size_t)N_NODES * 4);
    float* ald2  = (float*)alloc((size_t)N_NODES * 4);
    float* den2  = (float*)alloc((size_t)N_NODES * 4);
    float* ew2   = (float*)alloc((size_t)EDGES * 4);
    (void)alloc(4096);  // slack (B-tile overshoot for N=5 stays in-bounds)
    (void)ws_size; (void)n_in; (void)in_sizes; (void)out_size;

    float* logits = (float*)d_out;
    float* featc  = (float*)d_out + (size_t)N_NODES * 5;

    // 0) convert + pack
    norm_convert_kernel<<<N_NODES / 8, 256, 0, stream>>>(x, xnbf, xbf);
    pack_weight_kernel<<<(DIM/2*3072 + 255)/256, 256, 0, stream>>>(W1,  W1p,  DIM, 3072);
    pack_weight_kernel<<<(DIM/2*3072 + 255)/256, 256, 0, stream>>>(Wr1, Wr1p, DIM, 3072);
    pack_weight_kernel<<<(3072/2*DIM + 255)/256, 256, 0, stream>>>(W2,  W2p,  3072, DIM);
    pack_weight_kernel<<<(3072/2*DIM + 255)/256, 256, 0, stream>>>(Wr2, Wr2p, 3072, DIM);
    pack_weight_kernel<<<(DIM/2*HID  + 255)/256, 256, 0, stream>>>(Wh1, Wh1p, DIM, HID);
    pack_weight_kernel<<<(DIM/2*128  + 255)/256, 256, 0, stream>>>(Wh2, Wh2p, DIM, 128);
    pack_weight_kernel<<<(DIM/2*5    + 255)/256, 256, 0, stream>>>(Wf,  Wfp,  DIM, 5);

    // 1) kNN edges via fused WMMA similarity + top-k
    sim_topk_kernel<<<N_NODES / 16, 256, 0, stream>>>(xnbf, nbr);

    // ---- GAT layer 1 (HEADS=4) ----
    gemm_bf16_kernel<<<dim3(3072 / 64, 64), 256, 0, stream>>>(
        xbf, W1p, nullptr, bufA, nullptr, N_NODES, 3072, DIM, 0);         // hlin1
    attn_logits_kernel<<<N_NODES * HEADS / 8, 256, 0, stream>>>(
        bufA, as1, ad1, als1, ald1, HEADS, HID);
    hipMemsetAsync(den1, 0, (size_t)N_NODES * HEADS * 4, stream);
    edge_weights_kernel<<<EDGES / 256, 256, 0, stream>>>(nbr, als1, ald1, ew1, den1, HEADS);
    hipMemsetAsync(bufB, 0, (size_t)N_NODES * 3072 * 4, stream);
    scatter_kernel<<<EDGES, 256, 0, stream>>>(nbr, ew1, den1, bufA, bufB, HEADS, HID);
    gemm_bf16_kernel<<<dim3(3072 / 64, 64), 256, 0, stream>>>(
        xbf, Wr1p, br1, bufA, nullptr, N_NODES, 3072, DIM, 0);            // xres1 (reuse bufA)
    epilogue_kernel<<<4096, 256, 0, stream>>>(bufB, b1, bufA, hbf, N_NODES * 3072, 3072);

    // ---- GAT layer 2 (1 head) ----
    float* hlin2 = bufA;
    float* xres2 = (float*)((char*)bufA + (size_t)50331648);
    gemm_bf16_kernel<<<dim3(768 / 64, 64), 256, 0, stream>>>(
        hbf, W2p, nullptr, hlin2, nullptr, N_NODES, 768, 3072, 0);
    attn_logits_kernel<<<N_NODES / 8, 256, 0, stream>>>(hlin2, as2, ad2, als2, ald2, 1, DIM);
    hipMemsetAsync(den2, 0, (size_t)N_NODES * 4, stream);
    edge_weights_kernel<<<EDGES / 256, 256, 0, stream>>>(nbr, als2, ald2, ew2, den2, 1);
    hipMemsetAsync(bufB, 0, (size_t)N_NODES * 768 * 4, stream);
    scatter_kernel<<<EDGES, 256, 0, stream>>>(nbr, ew2, den2, hlin2, bufB, 1, DIM);
    gemm_bf16_kernel<<<dim3(768 / 64, 64), 256, 0, stream>>>(
        hbf, Wr2p, br2, xres2, nullptr, N_NODES, 768, 3072, 0);
    u16* featbf = xbf;  // reuse
    epilogue_kernel<<<4096, 256, 0, stream>>>(bufB, b2, xres2, featbf, N_NODES * 768, 768);

    // ---- heads ----
    gemm_bf16_kernel<<<dim3(1, 64), 256, 0, stream>>>(
        featbf, Wfp, bf, logits, nullptr, N_NODES, 5, DIM, 0);            // logits
    u16* tbf = xnbf;    // reuse
    gemm_bf16_kernel<<<dim3(768 / 64, 64), 256, 0, stream>>>(
        featbf, Wh1p, bh1, nullptr, tbf, N_NODES, 768, DIM, 1);           // relu MLP hidden
    gemm_bf16_kernel<<<dim3(2, 64), 256, 0, stream>>>(
        tbf, Wh2p, bh2, featc, nullptr, N_NODES, 128, DIM, 0);            // feat_c raw
    l2norm_kernel<<<N_NODES / 8, 256, 0, stream>>>(featc, 128);
}
